// FixedSlotAttentionMultiHead_50216757625630
// MI455X (gfx1250) — matrix operand order repaired and checked by reading, not compiled, not run on hardware
//
#include <hip/hip_runtime.h>
#include <stdint.h>

#define BB 16
#define NN 4096
#define DD 256
#define SS 8
#define HH 256
#define ITERS 3
#define QK_PAD 16

static_assert(sizeof(__bf16) == 2, "bf16");

typedef __attribute__((ext_vector_type(16))) __bf16 v16bf;
typedef __attribute__((ext_vector_type(8)))  __bf16 v8bf;
typedef __attribute__((ext_vector_type(8)))  float  v8f;

__device__ __forceinline__ float blockReduceSum256(float v, float* red) {
    int t = threadIdx.x;
    red[t] = v;
    __syncthreads();
    #pragma unroll
    for (int s = 128; s > 0; s >>= 1) {
        if (t < s) red[t] += red[t + s];
        __syncthreads();
    }
    float r = red[0];
    __syncthreads();
    return r;
}

// ---------------- phase 0 ----------------

__global__ void k_zero_bf16(__bf16* p, int n) {
    int i = blockIdx.x * 256 + threadIdx.x;
    if (i < n) p[i] = (__bf16)0.0f;
}

// slots = mu + exp(logsigma) * noise   ([1,S,D] broadcast over B)
__global__ void k_slots_init(const float* mu, const float* ls, const float* noise, float* slots) {
    int i = blockIdx.x * 256 + threadIdx.x;           // B*S*D = 32768
    int j = i & (SS * DD - 1);
    slots[i] = mu[j] + expf(ls[j]) * noise[i];
}

// LayerNorm(embeddings) -> bf16, one row (D=256) per block
__global__ void k_ln_emb(const float* __restrict__ x, const float* __restrict__ g,
                         const float* __restrict__ be, __bf16* __restrict__ out) {
    __shared__ float red[256];
    size_t row = blockIdx.x;
    int t = threadIdx.x;
    float v  = x[row * DD + t];
    float s  = blockReduceSum256(v, red);
    float mu = s * (1.0f / DD);
    float d  = v - mu;
    float s2 = blockReduceSum256(d * d, red);
    float inv = rsqrtf(s2 * (1.0f / DD) + 1e-5f);
    out[row * DD + t] = (__bf16)(d * inv * g[t] + be[t]);
}

// batched transpose: dst[bt][c][r] = src[bt][r][c]
__global__ void k_transpose(const float* __restrict__ src, float* __restrict__ dst, int R, int C) {
    int bt = blockIdx.y;
    int i = blockIdx.x * 256 + threadIdx.x;
    if (i >= R * C) return;
    int r = i / C, c = i - r * C;
    dst[(size_t)bt * R * C + (size_t)c * R + r] = src[(size_t)bt * R * C + (size_t)r * C + c];
}

// ---------------- per-iteration ----------------

// LN(slots) -> q = sn @ WqT -> qk[d] = sum_e q[e]*Wk[s,e,d], scaled, bf16, padded to 16 slot rows
__global__ void k_qk(const float* __restrict__ slots, const float* __restrict__ g_s,
                     const float* __restrict__ be_s, const float* __restrict__ WqT,
                     const float* __restrict__ Wk, __bf16* __restrict__ qkpad) {
    __shared__ float red[256], sn[256], q[256];
    int b = blockIdx.x >> 3, s = blockIdx.x & 7, t = threadIdx.x;
    float v  = slots[((size_t)(b * SS + s)) * DD + t];
    float su = blockReduceSum256(v, red);
    float mu = su * (1.0f / DD);
    float d  = v - mu;
    float s2 = blockReduceSum256(d * d, red);
    float inv = rsqrtf(s2 * (1.0f / DD) + 1e-5f);
    sn[t] = d * inv * g_s[t] + be_s[t];
    __syncthreads();
    float qa = 0.f;
    #pragma unroll 4
    for (int e = 0; e < DD; ++e) qa += sn[e] * WqT[e * DD + t];
    q[t] = qa;
    __syncthreads();
    const float* wk = Wk + (size_t)s * DD * DD;
    float acc = 0.f;
    #pragma unroll 4
    for (int e = 0; e < DD; ++e) acc += q[e] * wk[e * DD + t];   // coalesced across t
    qkpad[((size_t)(b * QK_PAD + s)) * DD + t] = (__bf16)(acc * 0.0625f);  // * D^-0.5
}

// WMMA: dots tile = emb(16x256 bf16) x qk^T(256x16 bf16), then softmax over slots per token.
// A fragments for a whole tile are preloaded (one load clause, one wait), then the 8
// WMMAs issue back-to-back. Each wave owns tile[wv] exclusively -> no block barrier;
// intra-wave DS ordering + compiler dscnt waits handle the cross-lane LDS handoff.
__global__ void __launch_bounds__(128) k_dots_softmax(const __bf16* __restrict__ emb,
                                                      const __bf16* __restrict__ qk,
                                                      float* __restrict__ attn) {
    __shared__ float tile[4][16][17];
    int b    = blockIdx.x >> 4;
    int tok0 = (blockIdx.x & 15) * 256;
    int wv   = threadIdx.x >> 5;
    int lane = threadIdx.x & 31;
    int l15  = lane & 15;
    int offA = (lane < 16) ? 0 : 8;    // C: row-half select (rows r+8 for lanes 16-31)
    int offB = (lane < 16) ? 0 : 16;   // B: K-half select
    const __bf16* qrow = qk + ((size_t)b * QK_PAD + l15) * DD;   // column l15 of B = slot row

    // B fragments are loop-invariant for the wave: keep all 8 resident.
    v16bf bfr[8];
    #pragma unroll
    for (int k = 0; k < 8; ++k)
        bfr[k] = *(const v16bf*)(qrow + k * 32 + offB);

    for (int tI = 0; tI < 4; ++tI) {
        int m0 = tok0 + wv * 64 + tI * 16;
        const __bf16* arow = emb + ((size_t)b * NN + m0 + l15) * DD;

        // Preload all A fragments for this tile (16x b128 loads, one wait).
        v8bf alo[8], ahi[8];
        #pragma unroll
        for (int k = 0; k < 8; ++k) {
            alo[k] = *(const v8bf*)(arow + k * 32 + offB);
            ahi[k] = *(const v8bf*)(arow + k * 32 + 8 + offB);
        }
        v8f acc = {};
        #pragma unroll
        for (int k = 0; k < 8; ++k) {
            v16bf a = __builtin_shufflevector(alo[k], ahi[k], 0, 1, 2, 3, 4, 5, 6, 7,
                                              8, 9, 10, 11, 12, 13, 14, 15);
            acc = __builtin_amdgcn_wmma_f32_16x16x32_bf16(
                false, a, false, bfr[k], (short)0, acc, false, false);
        }
        // C layout: VGPR r -> row M=r (lanes 0-15) / M=r+8 (lanes 16-31), col = lane&15
        #pragma unroll
        for (int r = 0; r < 8; ++r) tile[wv][r + offA][l15] = acc[r];
        if (lane < 16) {
            float vals[SS], m = -1e30f;
            #pragma unroll
            for (int s = 0; s < SS; ++s) { vals[s] = tile[wv][lane][s]; m = fmaxf(m, vals[s]); }
            float e[SS], sum = 0.f;
            #pragma unroll
            for (int s = 0; s < SS; ++s) { e[s] = expf(vals[s] - m); sum += e[s]; }
            float isum = 1.0f / sum;
            int n = m0 + lane;
            #pragma unroll
            for (int s = 0; s < SS; ++s)
                attn[((size_t)b * SS + s) * NN + n] = e[s] * isum;   // [B,S,N], coalesced
        }
    }
}

// per (b,s): sum of attn over tokens
__global__ void k_attn_sum(const float* __restrict__ attn, float* __restrict__ asum) {
    __shared__ float red[256];
    int bs = blockIdx.x, t = threadIdx.x;
    const float* p = attn + (size_t)bs * NN;
    float a = 0.f;
    for (int n = t; n < NN; n += 256) a += p[n];
    float s = blockReduceSum256(a, red);
    if (t == 0) asum[bs] = s;
}

// partial pooled vectors: part[c][b][s][e] = sum over 256-token chunk of attn_w * emb
__global__ void k_u0_part(const float* __restrict__ attn, const float* __restrict__ asum,
                          const __bf16* __restrict__ emb, float* __restrict__ part) {
    __shared__ float w[SS][256];
    __shared__ float isum[SS];
    int b = blockIdx.x >> 4, c = blockIdx.x & 15, t = threadIdx.x;
    int n0 = c * 256;
    if (t < SS) isum[t] = 1.0f / asum[b * SS + t];
    __syncthreads();
    #pragma unroll
    for (int s = 0; s < SS; ++s)
        w[s][t] = attn[((size_t)b * SS + s) * NN + n0 + t] * isum[s];
    __syncthreads();
    float acc[SS];
    #pragma unroll
    for (int s = 0; s < SS; ++s) acc[s] = 0.f;
    const __bf16* ep = emb + ((size_t)b * NN + n0) * DD + t;
    for (int n = 0; n < 256; ++n) {
        float em = (float)ep[(size_t)n * DD];   // coalesced across t
        #pragma unroll
        for (int s = 0; s < SS; ++s) acc[s] += w[s][n] * em;
    }
    #pragma unroll
    for (int s = 0; s < SS; ++s)
        part[(((size_t)c * BB + b) * SS + s) * DD + t] = acc[s];
}

__global__ void k_u0_reduce(const float* __restrict__ part, float* __restrict__ u0) {
    int i = blockIdx.x * 256 + threadIdx.x;   // 32768
    float a = 0.f;
    #pragma unroll
    for (int c = 0; c < 16; ++c) a += part[(size_t)c * (BB * SS * DD) + i];
    u0[i] = a;
}

// updates = u0 @ Wv^T ; GRU ; LN ; MLP ; residual
__global__ void k_gru_mlp(const float* __restrict__ u0, const float* __restrict__ WvT,
                          const float* __restrict__ slots_prev,
                          const float* __restrict__ WihT, const float* __restrict__ WhhT,
                          const float* __restrict__ b_ih, const float* __restrict__ b_hh,
                          const float* __restrict__ W1T, const float* __restrict__ b1,
                          const float* __restrict__ W2T, const float* __restrict__ b2,
                          const float* __restrict__ g_ff, const float* __restrict__ be_ff,
                          float* __restrict__ slots_out) {
    __shared__ float red[256], u0r[256], x[256], hp[256], hbuf[256], ff[256], hid[256];
    int b = blockIdx.x >> 3, s = blockIdx.x & 7, t = threadIdx.x;
    size_t row = (size_t)(b * SS + s) * DD;
    u0r[t] = u0[row + t];
    hp[t]  = slots_prev[row + t];
    __syncthreads();
    const float* wv = WvT + (size_t)s * DD * DD;
    float xa = 0.f;
    #pragma unroll 4
    for (int e = 0; e < DD; ++e) xa += u0r[e] * wv[e * DD + t];
    x[t] = xa;
    __syncthreads();
    float ar = b_ih[t], az = b_ih[256 + t], an = b_ih[512 + t];
    float hr = b_hh[t], hz = b_hh[256 + t], hn = b_hh[512 + t];
    #pragma unroll 2
    for (int d = 0; d < DD; ++d) {
        float xv = x[d], hv = hp[d];
        const float* wi = WihT + (size_t)d * 768;
        const float* wh = WhhT + (size_t)d * 768;
        ar += xv * wi[t];       az += xv * wi[256 + t]; an += xv * wi[512 + t];
        hr += hv * wh[t];       hz += hv * wh[256 + t]; hn += hv * wh[512 + t];
    }
    float r  = 1.0f / (1.0f + expf(-(ar + hr)));
    float z  = 1.0f / (1.0f + expf(-(az + hz)));
    float nn = tanhf(an + r * hn);
    float h  = (1.0f - z) * nn + z * hp[t];
    hbuf[t] = h;
    __syncthreads();
    float su = blockReduceSum256(h, red);
    float mu = su * (1.0f / DD);
    float dd = h - mu;
    float s2 = blockReduceSum256(dd * dd, red);
    float inv = rsqrtf(s2 * (1.0f / DD) + 1e-5f);
    ff[t] = dd * inv * g_ff[t] + be_ff[t];
    __syncthreads();
    float ha = b1[t];
    #pragma unroll 4
    for (int d = 0; d < DD; ++d) ha += ff[d] * W1T[d * HH + t];
    hid[t] = fmaxf(ha, 0.0f);
    __syncthreads();
    float oa = b2[t];
    #pragma unroll 4
    for (int hh = 0; hh < HH; ++hh) oa += hid[hh] * W2T[hh * DD + t];
    slots_out[row + t] = hbuf[t] + oa;
}

// ---------------- launch ----------------

extern "C" void kernel_launch(void* const* d_in, const int* in_sizes, int n_in,
                              void* d_out, int out_size, void* d_ws, size_t ws_size,
                              hipStream_t stream) {
    (void)in_sizes; (void)n_in; (void)out_size; (void)ws_size;
    const float* embeddings = (const float*)d_in[0];
    const float* noise      = (const float*)d_in[1];
    const float* mu         = (const float*)d_in[2];
    const float* logsig     = (const float*)d_in[3];
    const float* Wk         = (const float*)d_in[4];
    const float* Wq         = (const float*)d_in[5];
    const float* Wv         = (const float*)d_in[6];
    const float* W_ih       = (const float*)d_in[7];
    const float* W_hh       = (const float*)d_in[8];
    const float* b_ih       = (const float*)d_in[9];
    const float* b_hh       = (const float*)d_in[10];
    const float* W1         = (const float*)d_in[11];
    const float* b1         = (const float*)d_in[12];
    const float* W2         = (const float*)d_in[13];
    const float* b2         = (const float*)d_in[14];
    const float* g_in       = (const float*)d_in[15];
    const float* be_in      = (const float*)d_in[16];
    const float* g_s        = (const float*)d_in[17];
    const float* be_s       = (const float*)d_in[18];
    const float* g_ff       = (const float*)d_in[19];
    const float* be_ff      = (const float*)d_in[20];

    uint8_t* wsp = (uint8_t*)d_ws;
    auto alloc = [&](size_t bytes) -> void* {
        void* p = wsp;
        wsp += (bytes + 255) & ~(size_t)255;
        return p;
    };
    __bf16* embb   = (__bf16*)alloc((size_t)BB * NN * DD * 2);
    __bf16* qkpad  = (__bf16*)alloc((size_t)BB * QK_PAD * DD * 2);
    float*  slots  = (float*) alloc((size_t)BB * SS * DD * 4);
    float*  attn_s = (float*) alloc((size_t)BB * SS * NN * 4);
    float*  asum   = (float*) alloc((size_t)BB * SS * 4);
    float*  u0part = (float*) alloc((size_t)16 * BB * SS * DD * 4);
    float*  u0     = (float*) alloc((size_t)BB * SS * DD * 4);
    float*  WqT    = (float*) alloc((size_t)DD * DD * 4);
    float*  WvT    = (float*) alloc((size_t)SS * DD * DD * 4);
    float*  WihT   = (float*) alloc((size_t)768 * DD * 4);
    float*  WhhT   = (float*) alloc((size_t)768 * DD * 4);
    float*  W1T    = (float*) alloc((size_t)HH * DD * 4);
    float*  W2T    = (float*) alloc((size_t)DD * HH * 4);

    float* out_slots = (float*)d_out;                 // [B,S,D]
    float* out_attn  = (float*)d_out + BB * SS * DD;  // [B,S,N]

    // phase 0
    k_zero_bf16<<<(BB * QK_PAD * DD + 255) / 256, 256, 0, stream>>>(qkpad, BB * QK_PAD * DD);
    k_slots_init<<<(BB * SS * DD) / 256, 256, 0, stream>>>(mu, logsig, noise, slots);
    k_ln_emb<<<BB * NN, 256, 0, stream>>>(embeddings, g_in, be_in, embb);
    k_transpose<<<dim3(256, 1), 256, 0, stream>>>(Wq, WqT, DD, DD);
    k_transpose<<<dim3(256, SS), 256, 0, stream>>>(Wv, WvT, DD, DD);
    k_transpose<<<dim3(768, 1), 256, 0, stream>>>(W_ih, WihT, 768, DD);
    k_transpose<<<dim3(768, 1), 256, 0, stream>>>(W_hh, WhhT, 768, DD);
    k_transpose<<<dim3(256, 1), 256, 0, stream>>>(W1, W1T, HH, DD);
    k_transpose<<<dim3(256, 1), 256, 0, stream>>>(W2, W2T, DD, HH);

    for (int it = 0; it < ITERS; ++it) {
        float* attn_p    = (it == ITERS - 1) ? out_attn : attn_s;
        float* slots_dst = (it == ITERS - 1) ? out_slots : slots;
        k_qk<<<BB * SS, 256, 0, stream>>>(slots, g_s, be_s, WqT, Wk, qkpad);
        k_dots_softmax<<<BB * 16, 128, 0, stream>>>(embb, qkpad, attn_p);
        k_attn_sum<<<BB * SS, 256, 0, stream>>>(attn_p, asum);
        k_u0_part<<<BB * 16, 256, 0, stream>>>(attn_p, asum, embb, u0part);
        k_u0_reduce<<<(BB * SS * DD) / 256, 256, 0, stream>>>(u0part, u0);
        k_gru_mlp<<<BB * SS, 256, 0, stream>>>(u0, WvT, slots, WihT, WhhT, b_ih, b_hh,
                                               W1T, b1, W2T, b2, g_ff, be_ff, slots_dst);
    }
}
